// Model_43636867727984
// MI455X (gfx1250) — matrix operand order repaired and checked by reading, not compiled, not run on hardware
//
#include <hip/hip_runtime.h>

#define S_LEN 256
#define BATCH 64
#define DMT 1024
#define DMA 100
#define DMAP 128 /* padded K for audio */
#define DMV 512
#define DG 512
#define HID 256
#define ROWS (S_LEN * BATCH) /* 16384 */
#define EPSV 1e-5f
#define BROW 72 /* LDS B-stage row stride in ushorts: 144B = 16B-aligned, bank-spread */

typedef __attribute__((ext_vector_type(16))) __bf16 v16bf;
typedef __attribute__((ext_vector_type(8))) float v8f;

union BV {
    unsigned u[8];
    v16bf v;
};

__device__ __forceinline__ unsigned short bf16_1(float x) {
    unsigned a = __float_as_uint(x);
    a = (a + 0x7FFFu + ((a >> 16) & 1u)) >> 16;
    return (unsigned short)a;
}
__device__ __forceinline__ float sigm(float x) { return 1.0f / (1.0f + __expf(-x)); }

// Workgroup barrier that fences ONLY LDS (DScnt): global loads/stores in flight keep draining
// in the background instead of being force-waited like __syncthreads()' full fence would.
__device__ __forceinline__ void lds_barrier() {
    asm volatile("s_wait_dscnt 0x0\n\t"
                 "s_barrier_signal -1\n\t"
                 "s_barrier_wait -1" ::: "memory");
}

// ---------------- LayerNorm statistics: mean/rstd per (tensor i, batch j) ----------------
__global__ void __launch_bounds__(256) ln_stats_kernel(const float* __restrict__ r1,
                                                       const float* __restrict__ r2,
                                                       const float* __restrict__ r3,
                                                       const float* __restrict__ r4,
                                                       float* __restrict__ stats /* [4][64][2] */) {
    const int i = blockIdx.x >> 6;
    const int j = blockIdx.x & 63;
    const float* R = (i == 0) ? r1 : (i == 1) ? r2 : (i == 2) ? r3 : r4;
    float s = 0.f, ss = 0.f;
    for (int t = 0; t < S_LEN; ++t) {
        const float* row = R + (size_t)t * BATCH * DMT + (size_t)j * DMT;
        for (int d = threadIdx.x; d < DMT; d += 256) {
            float x = row[d];
            s += x;
            ss += x * x;
        }
    }
    __shared__ float red0[256], red1[256];
    red0[threadIdx.x] = s;
    red1[threadIdx.x] = ss;
    __syncthreads();
    for (int st = 128; st > 0; st >>= 1) {
        if (threadIdx.x < st) {
            red0[threadIdx.x] += red0[threadIdx.x + st];
            red1[threadIdx.x] += red1[threadIdx.x + st];
        }
        __syncthreads();
    }
    if (threadIdx.x == 0) {
        const float n = (float)S_LEN * (float)DMT;
        float m = red0[0] / n;
        float v = red1[0] / n - m * m;
        stats[(i * BATCH + j) * 2 + 0] = m;
        stats[(i * BATCH + j) * 2 + 1] = rsqrtf(v + EPSV);
    }
}

// ---------------- U = mean of 4 normalized tensors, stored bf16 (GEMM A operand) ----------------
__global__ void __launch_bounds__(256) compute_u_kernel(const float* __restrict__ r1,
                                                        const float* __restrict__ r2,
                                                        const float* __restrict__ r3,
                                                        const float* __restrict__ r4,
                                                        const float* __restrict__ stats,
                                                        unsigned short* __restrict__ U) {
    const size_t idx = (size_t)blockIdx.x * 256 + threadIdx.x;
    if (idx >= (size_t)ROWS * DMT) return;
    const int j = (int)((idx >> 10) & 63); // (idx / 1024) % 64
    float u = (r1[idx] - stats[(0 * BATCH + j) * 2]) * stats[(0 * BATCH + j) * 2 + 1] +
              (r2[idx] - stats[(1 * BATCH + j) * 2]) * stats[(1 * BATCH + j) * 2 + 1] +
              (r3[idx] - stats[(2 * BATCH + j) * 2]) * stats[(2 * BATCH + j) * 2 + 1] +
              (r4[idx] - stats[(3 * BATCH + j) * 2]) * stats[(3 * BATCH + j) * 2 + 1];
    U[idx] = bf16_1(u * 0.25f);
}

// ---------------- f32 [R,Ks] -> bf16 [R,Kd] with zero padding (Kd >= Ks) ----------------
__global__ void __launch_bounds__(256) cvt_pad_bf16_kernel(const float* __restrict__ src,
                                                           unsigned short* __restrict__ dst,
                                                           int R, int Ks, int Kd) {
    const size_t idx = (size_t)blockIdx.x * 256 + threadIdx.x;
    if (idx >= (size_t)R * Kd) return;
    const int r = (int)(idx / Kd), k = (int)(idx % Kd);
    dst[idx] = (k < Ks) ? bf16_1(src[(size_t)r * Ks + k]) : (unsigned short)0;
}

// ---------------- GEMM: C[M,N] = A[M,K] @ B[N,K]^T + bias ----------------
// A, B bf16 row-major, lda=K, ldb=K, K multiple of 64, M multiple of 128.
// Block = 128x64 macro-tile: 8 waves x (16x64 tile, 4 accumulators). The 64x64 B chunk is
// staged in LDS, double-buffered with a register-carried software pipeline: iteration i
// computes from buf[cur], stores the chunk (i+1) held in registers into buf[cur^1], then
// issues the load of chunk (i+2). The DS-only barrier lets that load span the barrier and
// drain under the next iteration's 8 WMMAs. Branch-free, EXEC all-ones throughout.
__global__ void __launch_bounds__(256) gemm_bf16_kernel(
    const unsigned short* __restrict__ A, const unsigned short* __restrict__ B,
    const float* __restrict__ bias, float* __restrict__ C, unsigned short* __restrict__ C16,
    int M, int N, int K, int ldc, int coff, int ldc16, int coff16) {
    __shared__ __align__(16) unsigned short bstage[2][64 * BROW]; // 2 x 9 KB
    const int wave = threadIdx.x >> 5, lane = threadIdx.x & 31;
    const int hi = lane >> 4, l16 = lane & 15;
    const int ntn = N >> 6;
    const int mb = blockIdx.x / ntn;
    const int nb = blockIdx.x % ntn;
    const int m0 = mb * 128 + wave * 16;
    const int n0 = nb * 64;
    const unsigned short* Arow = A + (size_t)(m0 + l16) * K;
    // cooperative B fill: thread -> (n-local, k-part): 64 rows x 2 uint4 (elements kp, kp+32)
    const int nl = threadIdx.x >> 2;      // 0..63
    const int kp = (threadIdx.x & 3) * 8; // element offset 0,8,16,24
    const unsigned short* Bg = B + (size_t)(n0 + nl) * K + kp;
    const int nch = K >> 6;
    const int kmax = (nch - 1) << 6;
    { // chunk 0 directly into buffer 0
        const uint4 p0 = *reinterpret_cast<const uint4*>(Bg);
        const uint4 p1 = *reinterpret_cast<const uint4*>(Bg + 32);
        unsigned short* Bw = &bstage[0][0] + nl * BROW + kp;
        *reinterpret_cast<uint4*>(Bw) = p0;
        *reinterpret_cast<uint4*>(Bw + 32) = p1;
    }
    // register-carried prefetch of chunk 1 (clamped)
    int kpf = (64 < kmax) ? 64 : kmax;
    uint4 pf0 = *reinterpret_cast<const uint4*>(Bg + kpf);
    uint4 pf1 = *reinterpret_cast<const uint4*>(Bg + kpf + 32);
    lds_barrier();

    v8f acc0 = {0.f, 0.f, 0.f, 0.f, 0.f, 0.f, 0.f, 0.f};
    v8f acc1 = acc0, acc2 = acc0, acc3 = acc0;
    int cur = 0;
    for (int i = 0; i < nch; ++i) {
        const int k0 = i << 6;
        BV av0, av1, bv;
        {
            const uint4 a0 = *reinterpret_cast<const uint4*>(Arow + k0 + hi * 8);
            const uint4 a1 = *reinterpret_cast<const uint4*>(Arow + k0 + 16 + hi * 8);
            const uint4 a2 = *reinterpret_cast<const uint4*>(Arow + k0 + 32 + hi * 8);
            const uint4 a3 = *reinterpret_cast<const uint4*>(Arow + k0 + 48 + hi * 8);
            av0.u[0] = a0.x; av0.u[1] = a0.y; av0.u[2] = a0.z; av0.u[3] = a0.w;
            av0.u[4] = a1.x; av0.u[5] = a1.y; av0.u[6] = a1.z; av0.u[7] = a1.w;
            av1.u[0] = a2.x; av1.u[1] = a2.y; av1.u[2] = a2.z; av1.u[3] = a2.w;
            av1.u[4] = a3.x; av1.u[5] = a3.y; av1.u[6] = a3.z; av1.u[7] = a3.w;
        }
        const unsigned short* bs = &bstage[cur][0];
#define DO_T(NN, AV, SS, ACC)                                                                     \
    {                                                                                             \
        const unsigned short* bp = bs + ((NN)*16 + l16) * BROW + (SS)*32 + hi * 8;                \
        const uint4 b0 = *reinterpret_cast<const uint4*>(bp);                                     \
        const uint4 b1 = *reinterpret_cast<const uint4*>(bp + 16);                                \
        bv.u[0] = b0.x; bv.u[1] = b0.y; bv.u[2] = b0.z; bv.u[3] = b0.w;                           \
        bv.u[4] = b1.x; bv.u[5] = b1.y; bv.u[6] = b1.z; bv.u[7] = b1.w;                           \
        ACC = __builtin_amdgcn_wmma_f32_16x16x32_bf16(false, AV.v, false, bv.v, (short)0, ACC,    \
                                                      false, false);                              \
    }
        DO_T(0, av0, 0, acc0)
        DO_T(1, av0, 0, acc1)
        DO_T(2, av0, 0, acc2)
        DO_T(3, av0, 0, acc3)
        DO_T(0, av1, 1, acc0)
        DO_T(1, av1, 1, acc1)
        DO_T(2, av1, 1, acc2)
        DO_T(3, av1, 1, acc3)
#undef DO_T
        { // land the register-carried chunk (i+1) in the other buffer (load done an iter ago)
            unsigned short* Bw = &bstage[cur ^ 1][0] + nl * BROW + kp;
            *reinterpret_cast<uint4*>(Bw) = pf0;
            *reinterpret_cast<uint4*>(Bw + 32) = pf1;
        }
        { // issue load of chunk (i+2); DS-only barrier lets it drain under next 8 WMMAs
            int k2 = (i + 2) << 6;
            k2 = (k2 < kmax) ? k2 : kmax;
            pf0 = *reinterpret_cast<const uint4*>(Bg + k2);
            pf1 = *reinterpret_cast<const uint4*>(Bg + k2 + 32);
        }
        lds_barrier();
        cur ^= 1;
    }
    float bn[4];
#pragma unroll
    for (int nn = 0; nn < 4; ++nn) bn[nn] = bias ? bias[n0 + nn * 16 + l16] : 0.f;
    if (C) {
#pragma unroll
        for (int r = 0; r < 8; ++r) {
            float* crow = C + (size_t)(m0 + r + hi * 8) * ldc + coff + n0 + l16;
            crow[0] = acc0[r] + bn[0];
            crow[16] = acc1[r] + bn[1];
            crow[32] = acc2[r] + bn[2];
            crow[48] = acc3[r] + bn[3];
        }
    }
    if (C16) {
#pragma unroll
        for (int r = 0; r < 8; ++r) {
            unsigned short* crow = C16 + (size_t)(m0 + r + hi * 8) * ldc16 + coff16 + n0 + l16;
            crow[0] = bf16_1(acc0[r] + bn[0]);
            crow[16] = bf16_1(acc1[r] + bn[1]);
            crow[32] = bf16_1(acc2[r] + bn[2]);
            crow[48] = bf16_1(acc3[r] + bn[3]);
        }
    }
}

// ---------------- LSTM recurrence: one block per direction; h in LDS (bf16), c in registers ----------------
// pre* = x @ Wih^T + b, precomputed [S*B, 4H] f32. Whh* bf16 [4H][H]. Optional f32/bf16 h outputs.
// DS-only barriers keep the per-step global h-stores OFF the sequential critical path.
__global__ void __launch_bounds__(1024) lstm_rec_kernel(
    const float* __restrict__ preF, const float* __restrict__ preB,
    const unsigned short* __restrict__ WhhF, const unsigned short* __restrict__ WhhB,
    float* __restrict__ out, int ldo, unsigned short* __restrict__ out16, int ldo16, int cbase) {
    const int dir = blockIdx.x;
    const float* pre = dir ? preB : preF;
    const unsigned short* Whh = dir ? WhhB : WhhF;
    const int co = cbase + dir * HID;
    __shared__ __align__(16) unsigned short hbuf[BATCH * HID]; // 32 KB of 320 KB WGP LDS
    for (int i = threadIdx.x; i < BATCH * HID; i += 1024) hbuf[i] = 0;
    const int wave = threadIdx.x >> 5, lane = threadIdx.x & 31;
    const int hi = lane >> 4, l16 = lane & 15;
    float creg[2][8], hreg[2][8];
#pragma unroll
    for (int q = 0; q < 2; ++q)
#pragma unroll
        for (int r = 0; r < 8; ++r) creg[q][r] = 0.f;
    lds_barrier();

    for (int s = 0; s < S_LEN; ++s) {
        const int t = dir ? (S_LEN - 1 - s) : s;
#pragma unroll
        for (int q = 0; q < 2; ++q) {
            const int id = wave * 2 + q;   // 64 tasks: 4 batch-tiles x 16 hidden-tiles
            const int m0 = (id & 3) << 4;  // batch tile origin
            const int u0 = (id >> 2) << 4; // hidden-unit tile origin
            v8f acc0 = {0.f, 0.f, 0.f, 0.f, 0.f, 0.f, 0.f, 0.f};
            v8f acc1 = acc0, acc2 = acc0, acc3 = acc0;
            for (int k0 = 0; k0 < HID; k0 += 32) {
                BV av, bv;
                const uint4 a0 =
                    *reinterpret_cast<const uint4*>(&hbuf[(m0 + l16) * HID + k0 + hi * 8]);
                const uint4 a1 =
                    *reinterpret_cast<const uint4*>(&hbuf[(m0 + l16) * HID + k0 + 16 + hi * 8]);
                av.u[0] = a0.x; av.u[1] = a0.y; av.u[2] = a0.z; av.u[3] = a0.w;
                av.u[4] = a1.x; av.u[5] = a1.y; av.u[6] = a1.z; av.u[7] = a1.w;
#define LOADB(g, ACC)                                                                             \
    {                                                                                             \
        const unsigned short* wr = Whh + (size_t)((g)*HID + u0 + l16) * HID + k0 + hi * 8;        \
        const uint4 b0 = *reinterpret_cast<const uint4*>(wr);                                     \
        const uint4 b1 = *reinterpret_cast<const uint4*>(wr + 16);                                \
        bv.u[0] = b0.x; bv.u[1] = b0.y; bv.u[2] = b0.z; bv.u[3] = b0.w;                           \
        bv.u[4] = b1.x; bv.u[5] = b1.y; bv.u[6] = b1.z; bv.u[7] = b1.w;                           \
        ACC = __builtin_amdgcn_wmma_f32_16x16x32_bf16(false, av.v, false, bv.v, (short)0, ACC,    \
                                                      false, false);                              \
    }
                LOADB(0, acc0)
                LOADB(1, acc1)
                LOADB(2, acc2)
                LOADB(3, acc3)
#undef LOADB
            }
#pragma unroll
            for (int r = 0; r < 8; ++r) {
                const int b = m0 + r + hi * 8;
                const size_t prow = (size_t)(t * BATCH + b) * (4 * HID);
                const int col = u0 + l16;
                float gi = acc0[r] + pre[prow + 0 * HID + col];
                float gf = acc1[r] + pre[prow + 1 * HID + col];
                float gg = acc2[r] + pre[prow + 2 * HID + col];
                float go = acc3[r] + pre[prow + 3 * HID + col];
                float c = sigm(gf) * creg[q][r] + sigm(gi) * tanhf(gg);
                creg[q][r] = c;
                hreg[q][r] = sigm(go) * tanhf(c);
            }
        }
        lds_barrier(); // all waves finished reading this step's h (DS only; stores keep draining)
#pragma unroll
        for (int q = 0; q < 2; ++q) {
            const int id = wave * 2 + q;
            const int m0 = (id & 3) << 4;
            const int u0 = (id >> 2) << 4;
#pragma unroll
            for (int r = 0; r < 8; ++r) {
                const int b = m0 + r + hi * 8;
                const float h = hreg[q][r];
                const unsigned short h16 = bf16_1(h);
                hbuf[b * HID + u0 + l16] = h16;
                if (out) out[(size_t)(t * BATCH + b) * ldo + co + u0 + l16] = h;
                if (out16) out16[(size_t)(t * BATCH + b) * ldo16 + co + u0 + l16] = h16;
            }
        }
        lds_barrier(); // publish h for next step
    }
}

// ---------------- Final ragged gather: [f_l | f_a | f_v | f_u | r2n | U_a | U_v] ----------------
__global__ void __launch_bounds__(256) gather_kernel(const int* __restrict__ nidx,
                                                     const float* __restrict__ elav,
                                                     const float* __restrict__ fusedb,
                                                     const float* __restrict__ r2,
                                                     const float* __restrict__ stats,
                                                     const float* __restrict__ U_a,
                                                     const float* __restrict__ U_v,
                                                     float* __restrict__ out) {
    const int n = blockIdx.x;
    const int flat = nidx[n];
    const int j = flat & 63;
    const float m2 = stats[(1 * BATCH + j) * 2];
    const float rs2 = stats[(1 * BATCH + j) * 2 + 1];
    float* o = out + (size_t)n * 3684;
    const float* el = elav + (size_t)flat * (3 * DG);
    for (int d = threadIdx.x; d < 3 * DG; d += 256) o[d] = el[d]; // l, a, v
    const float* fu = fusedb + (size_t)flat * DG;
    for (int d = threadIdx.x; d < DG; d += 256) o[3 * DG + d] = fu[d];
    const float* rr = r2 + (size_t)flat * DMT;
    for (int d = threadIdx.x; d < DMT; d += 256) o[2048 + d] = (rr[d] - m2) * rs2;
    const float* ua = U_a + (size_t)flat * DMA;
    for (int d = threadIdx.x; d < DMA; d += 256) o[3072 + d] = ua[d];
    const float* uv = U_v + (size_t)flat * DMV;
    for (int d = threadIdx.x; d < DMV; d += 256) o[3172 + d] = uv[d];
}

extern "C" void kernel_launch(void* const* d_in, const int* in_sizes, int n_in, void* d_out,
                              int out_size, void* d_ws, size_t ws_size, hipStream_t stream) {
    (void)n_in;
    (void)out_size;
    (void)ws_size;
    const float* r1 = (const float*)d_in[0];
    const float* r2 = (const float*)d_in[1];
    const float* r3 = (const float*)d_in[2];
    const float* r4 = (const float*)d_in[3];
    const float* U_a = (const float*)d_in[5];
    const float* U_v = (const float*)d_in[6];
    const float* W_a = (const float*)d_in[7];
    const float* b_a = (const float*)d_in[8];
    const float* W_v = (const float*)d_in[9];
    const float* b_v = (const float*)d_in[10];
    const float* W_l = (const float*)d_in[11];
    const float* b_l = (const float*)d_in[12];
    const float* Wih = (const float*)d_in[13]; // [2][2][1024][512]
    const float* Whh = (const float*)d_in[14]; // [2][2][1024][256]
    const float* lb = (const float*)d_in[15];  // [2][2][1024]
    const float* W_u = (const float*)d_in[16];
    const float* b_u = (const float*)d_in[17];
    const int* nidx = (const int*)d_in[18];
    const int Nn = in_sizes[18];
    float* out = (float*)d_out;

    char* w = (char*)d_ws;
    size_t off = 0;
    auto take = [&](size_t bytes) -> void* {
        void* p = w + off;
        off += (bytes + 255) & ~(size_t)255;
        return p;
    };
    float* stats = (float*)take((size_t)4 * BATCH * 2 * sizeof(float));
    unsigned short* U16 = (unsigned short*)take((size_t)ROWS * DMT * 2);
    unsigned short* Ua16 = (unsigned short*)take((size_t)ROWS * DMAP * 2);
    unsigned short* Uv16 = (unsigned short*)take((size_t)ROWS * DMV * 2);
    unsigned short* Wa16 = (unsigned short*)take((size_t)DG * DMAP * 2);
    unsigned short* Wv16 = (unsigned short*)take((size_t)DG * DMV * 2);
    unsigned short* Wl16 = (unsigned short*)take((size_t)DG * DMT * 2);
    unsigned short* Wih16 = (unsigned short*)take((size_t)4 * 4 * HID * DG * 2);
    unsigned short* Whh16 = (unsigned short*)take((size_t)4 * 4 * HID * HID * 2);
    unsigned short* Wu16 = (unsigned short*)take((size_t)DG * 3 * DG * 2);
    unsigned short* xproj16 = (unsigned short*)take((size_t)ROWS * DG * 2);
    unsigned short* x116 = (unsigned short*)take((size_t)ROWS * DG * 2);
    float* elav = (float*)take((size_t)ROWS * 3 * DG * 4); // [l | a | v], ld = 1536
    unsigned short* elav16 = (unsigned short*)take((size_t)ROWS * 3 * DG * 2);
    float* fusedb = (float*)take((size_t)ROWS * DG * 4);
    float* pre0 = (float*)take((size_t)ROWS * 4 * HID * 4);
    float* pre1 = (float*)take((size_t)ROWS * 4 * HID * 4);

    auto cvt = [&](const float* src, unsigned short* dst, int R, int Ks, int Kd) {
        size_t total = (size_t)R * Kd;
        cvt_pad_bf16_kernel<<<dim3((unsigned)((total + 255) / 256)), dim3(256), 0, stream>>>(
            src, dst, R, Ks, Kd);
    };
    // 1) LayerNorm stats + U (bf16)
    ln_stats_kernel<<<dim3(4 * BATCH), dim3(256), 0, stream>>>(r1, r2, r3, r4, stats);
    compute_u_kernel<<<dim3((unsigned)(((size_t)ROWS * DMT + 255) / 256)), dim3(256), 0, stream>>>(
        r1, r2, r3, r4, stats, U16);
    // 2) one-time bf16 conversions (all GEMM operands become guard-free aligned b128 loads)
    cvt(U_a, Ua16, ROWS, DMA, DMAP);
    cvt(U_v, Uv16, ROWS, DMV, DMV);
    cvt(W_a, Wa16, DG, DMA, DMAP);
    cvt(W_v, Wv16, DG, DMV, DMV);
    cvt(W_l, Wl16, DG, DMT, DMT);
    cvt(Wih, Wih16, 4 * 4 * HID, DG, DG);
    cvt(Whh, Whh16, 4 * 4 * HID, HID, HID);
    cvt(W_u, Wu16, DG, 3 * DG, 3 * DG);

    auto gemm = [&](const unsigned short* A, const unsigned short* B, const float* bias, float* C,
                    unsigned short* C16, int N, int K, int ldc, int coff, int ldc16, int coff16) {
        int blocks = (ROWS / 128) * (N / 64); // exact: no tail
        gemm_bf16_kernel<<<dim3(blocks), dim3(256), 0, stream>>>(A, B, bias, C, C16, ROWS, N, K,
                                                                 ldc, coff, ldc16, coff16);
    };
    // 3) emotions_a / emotions_v -> elav (f32) + elav16 (bf16) columns [512..1024) / [1024..1536)
    gemm(Ua16, Wa16, b_a, elav, elav16, DG, DMAP, 3 * DG, DG, 3 * DG, DG);
    gemm(Uv16, Wv16, b_v, elav, elav16, DG, DMV, 3 * DG, 2 * DG, 3 * DG, 2 * DG);
    // 4) xproj = U @ W_l^T + b_l (only consumed as GEMM A -> bf16 only)
    gemm(U16, Wl16, b_l, nullptr, xproj16, DG, DMT, 0, 0, DG, 0);
    // 5) LSTM layer 0: gate projections, then recurrence -> x1 bf16 (fwd cols 0..255, bwd 256..511)
    gemm(xproj16, Wih16 + (size_t)0 * 4 * HID * DG, lb + 0 * 4 * HID, pre0, nullptr, 4 * HID, DG,
         4 * HID, 0, 0, 0);
    gemm(xproj16, Wih16 + (size_t)1 * 4 * HID * DG, lb + 1 * 4 * HID, pre1, nullptr, 4 * HID, DG,
         4 * HID, 0, 0, 0);
    lstm_rec_kernel<<<dim3(2), dim3(1024), 0, stream>>>(
        pre0, pre1, Whh16 + (size_t)0 * 4 * HID * HID, Whh16 + (size_t)1 * 4 * HID * HID, nullptr,
        0, x116, DG, 0);
    // 6) LSTM layer 1 -> emotions_l region of elav + elav16 (cols 0..511)
    gemm(x116, Wih16 + (size_t)2 * 4 * HID * DG, lb + 2 * 4 * HID, pre0, nullptr, 4 * HID, DG,
         4 * HID, 0, 0, 0);
    gemm(x116, Wih16 + (size_t)3 * 4 * HID * DG, lb + 3 * 4 * HID, pre1, nullptr, 4 * HID, DG,
         4 * HID, 0, 0, 0);
    lstm_rec_kernel<<<dim3(2), dim3(1024), 0, stream>>>(
        pre0, pre1, Whh16 + (size_t)2 * 4 * HID * HID, Whh16 + (size_t)3 * 4 * HID * HID, elav,
        3 * DG, elav16, 3 * DG, 0);
    // 7) fused = [l|a|v] @ W_u^T + b_u (bf16 A, K=1536)
    gemm(elav16, Wu16, b_u, fusedb, nullptr, DG, 3 * DG, DG, 0, 0, 0);
    // 8) ragged gather into output [N, 3684]
    gather_kernel<<<dim3(Nn), dim3(256), 0, stream>>>(nidx, elav, fusedb, r2, stats, U_a, U_v,
                                                      out);
}
